// HeteroGraphConv_77489799954569
// MI455X (gfx1250) — matrix operand order repaired and checked by reading, not compiled, not run on hardware
//
#include <hip/hip_runtime.h>
#include <stdint.h>

typedef __attribute__((ext_vector_type(16))) __bf16 v16bf;
typedef __attribute__((ext_vector_type(8)))  float  v8f;

#define HIDDEN 256
#define ECNT 40000

__device__ __forceinline__ unsigned short f2bf(float f) {
    unsigned int u = __float_as_uint(f);
    unsigned int r = u + 0x7fffu + ((u >> 16) & 1u);   // round-to-nearest-even
    return (unsigned short)(r >> 16);
}

union FragU {
    uint4 u[2];
    unsigned short h[16];
    v16bf v;
};

#define WMMA_BF16(A, B, C) \
    __builtin_amdgcn_wmma_f32_16x16x32_bf16(false, (A), false, (B), (short)0, (C), false, false)

// ---------------------------------------------------------------------------
// Softmax of per-node-type edge-type weights -> one scalar per relation
// rel order (EDGE_TYPES): 0 q2text 1 q2table 2 q2cell 3 q2image 4 tbl_cell
//                         5 txt_tbl 6 txt_img 7 c2c_row 8 c2c_col 9 img_cap
// ---------------------------------------------------------------------------
__global__ void relw_kernel(const float* __restrict__ etw_caption,
                            const float* __restrict__ etw_cell,
                            const float* __restrict__ etw_image,
                            const float* __restrict__ etw_table,
                            const float* __restrict__ etw_textblock,
                            float* __restrict__ relw) {
    if (threadIdx.x != 0 || blockIdx.x != 0) return;
    relw[0] = 1.0f;              // textblock: single incoming rel
    relw[9] = 1.0f;              // caption: single incoming rel
    (void)etw_caption; (void)etw_textblock;
    {   // table: [query_to_table, text_refers_table] -> rel1, rel5
        float a = etw_table[0], b = etw_table[1];
        float m = fmaxf(a, b);
        float ea = __expf(a - m), eb = __expf(b - m);
        float s = ea + eb;
        relw[1] = ea / s; relw[5] = eb / s;
    }
    {   // image: [query_to_image, text_refers_image] -> rel3, rel6
        float a = etw_image[0], b = etw_image[1];
        float m = fmaxf(a, b);
        float ea = __expf(a - m), eb = __expf(b - m);
        float s = ea + eb;
        relw[3] = ea / s; relw[6] = eb / s;
    }
    {   // cell: [q2cell, table_contains_cell, c2c_row, c2c_col] -> rel2,4,7,8
        float e0 = etw_cell[0], e1 = etw_cell[1], e2 = etw_cell[2], e3 = etw_cell[3];
        float m = fmaxf(fmaxf(e0, e1), fmaxf(e2, e3));
        float a = __expf(e0 - m), b = __expf(e1 - m), c = __expf(e2 - m), d = __expf(e3 - m);
        float s = a + b + c + d;
        relw[2] = a / s; relw[4] = b / s; relw[7] = c / s; relw[8] = d / s;
    }
}

// ---------------------------------------------------------------------------
// fp32 -> bf16 elementwise (node features). One thread = 8 elements.
// ---------------------------------------------------------------------------
__global__ void xcvt_kernel(const float* __restrict__ x,
                            unsigned short* __restrict__ xb, int groups) {
    int i = blockIdx.x * blockDim.x + threadIdx.x;
    if (i >= groups) return;
    const float4* p = (const float4*)x + (size_t)i * 2;
    float4 a = p[0], b = p[1];
    FragU f;
    f.h[0] = f2bf(a.x); f.h[1] = f2bf(a.y); f.h[2] = f2bf(a.z); f.h[3] = f2bf(a.w);
    f.h[4] = f2bf(b.x); f.h[5] = f2bf(b.y); f.h[6] = f2bf(b.z); f.h[7] = f2bf(b.w);
    *(uint4*)(xb + (size_t)i * 8) = f.u[0];
}

// ---------------------------------------------------------------------------
// Re-layout one 256x256 fp32 weight (row-major, K-major: W[k][n]) into
// bf16 WMMA B-fragments: frag[kt*16+nt] is 1KB, lane-major, 16 halfs/lane.
// B k order used here (contiguous, same as A): lanes 0-15 hold K=j (j=0..15), lanes 16-31 hold K=16+j.
// grid = 128 fragments, block = 32 lanes.
// ---------------------------------------------------------------------------
__global__ void wfrag_kernel(const float* __restrict__ W,
                             unsigned short* __restrict__ frag) {
    int fragIdx = blockIdx.x;            // kt*16 + nt
    int kt = fragIdx >> 4;
    int nt = fragIdx & 15;
    int lane = threadIdx.x;
    int n = nt * 16 + (lane & 15);
    int kbase = kt * 32 + ((lane & 16) ? 16 : 0);
    FragU f;
#pragma unroll
    for (int j = 0; j < 16; ++j)
        f.h[j] = f2bf(W[(size_t)(kbase + j) * HIDDEN + n]);
    unsigned short* dst = frag + (size_t)fragIdx * 512 + lane * 16;
    *(uint4*)dst = f.u[0];
    *(uint4*)(dst + 8) = f.u[1];
}

// ---------------------------------------------------------------------------
// Per-destination in-degree
// ---------------------------------------------------------------------------
__global__ void deg_kernel(const int* __restrict__ dst,
                           float* __restrict__ deg, int E) {
    int i = blockIdx.x * blockDim.x + threadIdx.x;
    if (i < E) atomicAdd(deg + dst[i], 1.0f);
}

// ---------------------------------------------------------------------------
// Edge kernel: block = 32 edges (two M-tiles), 8 waves.
// Wave w owns head w (n = 32w..32w+31). Each B-fragment load feeds 2 WMMAs.
// Computes q,k,v via WMMA (bf16 x bf16 -> f32), per-head gate, atomic scatter.
// ---------------------------------------------------------------------------
__launch_bounds__(256)
__global__ void edge_kernel(const unsigned short* __restrict__ xb,
                            const int* __restrict__ src,
                            const int* __restrict__ dst,
                            const unsigned short* __restrict__ fq,
                            const unsigned short* __restrict__ fk,
                            const unsigned short* __restrict__ fv,
                            const float* __restrict__ bq,
                            const float* __restrict__ bk,
                            const float* __restrict__ bv,
                            const float* __restrict__ eattn,
                            const float* __restrict__ deg,
                            const float* __restrict__ relw,
                            float* __restrict__ agg, int E) {
    __shared__ __align__(16) unsigned short sA[32 * HIDDEN];
    __shared__ int   sDst[32];
    __shared__ float sDeg[32];

    int tid = threadIdx.x;
    int e0 = blockIdx.x * 32;

    if (tid < 32) {
        int e = e0 + tid; if (e >= E) e = E - 1;
        int d = dst[e];
        sDst[tid] = d;
        sDeg[tid] = fmaxf(deg[d], 1e-6f);
    }
    {   // gather 32 source rows (bf16) into LDS: thread t -> row t>>3, 32 halfs
        int e = e0 + (tid >> 3); if (e >= E) e = E - 1;
        int srow = src[e];
        const uint4* p = (const uint4*)(xb + (size_t)srow * HIDDEN + (tid & 7) * 32);
        uint4* q = (uint4*)(sA + (tid >> 3) * HIDDEN + (tid & 7) * 32);
        q[0] = p[0]; q[1] = p[1]; q[2] = p[2]; q[3] = p[3];
    }
    __syncthreads();

    int lane = tid & 31;
    int wave = tid >> 5;
    int n0 = wave * 2, n1 = n0 + 1;     // n-tiles owned by this wave (head==wave)
    int lg = (lane & 16) ? 8 : 0;       // C/D row-group offset
    int ln = lane & 15;

    v8f zero = {0, 0, 0, 0, 0, 0, 0, 0};
    v8f aq[2][2], ak[2][2], av[2][2];   // [mtile][ntile]
#pragma unroll
    for (int m = 0; m < 2; ++m)
#pragma unroll
        for (int t = 0; t < 2; ++t) { aq[m][t] = zero; ak[m][t] = zero; av[m][t] = zero; }

#pragma unroll
    for (int kt = 0; kt < 8; ++kt) {
        // A fragments (contiguous k, same order as B frags): lanes 0-15 hold K 0..15, 16-31 hold K 16..31
        FragU A0, A1;
        const unsigned short* ap0 = sA + ln * HIDDEN + kt * 32 + ((lane & 16) ? 16 : 0);
        A0.u[0] = *(const uint4*)ap0;
        A0.u[1] = *(const uint4*)(ap0 + 8);
        const unsigned short* ap1 = ap0 + 16 * HIDDEN;
        A1.u[0] = *(const uint4*)ap1;
        A1.u[1] = *(const uint4*)(ap1 + 8);

        size_t f0 = (size_t)(kt * 16 + n0) * 512 + lane * 16;
        size_t f1 = (size_t)(kt * 16 + n1) * 512 + lane * 16;
        FragU Bq0, Bq1, Bk0, Bk1, Bv0, Bv1;
        Bq0.u[0] = *(const uint4*)(fq + f0); Bq0.u[1] = *(const uint4*)(fq + f0 + 8);
        Bq1.u[0] = *(const uint4*)(fq + f1); Bq1.u[1] = *(const uint4*)(fq + f1 + 8);
        Bk0.u[0] = *(const uint4*)(fk + f0); Bk0.u[1] = *(const uint4*)(fk + f0 + 8);
        Bk1.u[0] = *(const uint4*)(fk + f1); Bk1.u[1] = *(const uint4*)(fk + f1 + 8);
        Bv0.u[0] = *(const uint4*)(fv + f0); Bv0.u[1] = *(const uint4*)(fv + f0 + 8);
        Bv1.u[0] = *(const uint4*)(fv + f1); Bv1.u[1] = *(const uint4*)(fv + f1 + 8);

        aq[0][0] = WMMA_BF16(A0.v, Bq0.v, aq[0][0]);
        aq[1][0] = WMMA_BF16(A1.v, Bq0.v, aq[1][0]);
        aq[0][1] = WMMA_BF16(A0.v, Bq1.v, aq[0][1]);
        aq[1][1] = WMMA_BF16(A1.v, Bq1.v, aq[1][1]);
        ak[0][0] = WMMA_BF16(A0.v, Bk0.v, ak[0][0]);
        ak[1][0] = WMMA_BF16(A1.v, Bk0.v, ak[1][0]);
        ak[0][1] = WMMA_BF16(A0.v, Bk1.v, ak[0][1]);
        ak[1][1] = WMMA_BF16(A1.v, Bk1.v, ak[1][1]);
        av[0][0] = WMMA_BF16(A0.v, Bv0.v, av[0][0]);
        av[1][0] = WMMA_BF16(A1.v, Bv0.v, av[1][0]);
        av[0][1] = WMMA_BF16(A0.v, Bv1.v, av[0][1]);
        av[1][1] = WMMA_BF16(A1.v, Bv1.v, av[1][1]);
    }

    float bq0v = bq[n0 * 16 + ln], bq1v = bq[n1 * 16 + ln];
    float bk0v = bk[n0 * 16 + ln], bk1v = bk[n1 * 16 + ln];
    float bv0v = bv[n0 * 16 + ln], bv1v = bv[n1 * 16 + ln];
    float rw = *relw;
    float ea = eattn[wave];
    const float inv_sqrt_d = 0.17677669529663687f;   // 1/sqrt(32)

#pragma unroll
    for (int m = 0; m < 2; ++m) {
#pragma unroll
        for (int r = 0; r < 8; ++r) {
            float q0 = aq[m][0][r] + bq0v, q1 = aq[m][1][r] + bq1v;
            float k0 = ak[m][0][r] + bk0v, k1 = ak[m][1][r] + bk1v;
            float p = q0 * k0 + q1 * k1;             // this lane's 2 dims of head dot
            p += __shfl_xor(p, 1);
            p += __shfl_xor(p, 2);
            p += __shfl_xor(p, 4);
            p += __shfl_xor(p, 8);                   // full 32-dim head dot per half-wave
            int eRow = m * 16 + r + lg;
            float att = p * inv_sqrt_d + ea;
            float w = fmaxf(att / sDeg[eRow], 0.0f);
            int e = e0 + eRow;
            if (e < E) {
                float s = rw * w;
                float* op = agg + (size_t)sDst[eRow] * HIDDEN;
                atomicAdd(op + n0 * 16 + ln, (av[m][0][r] + bv0v) * s);
                atomicAdd(op + n1 * 16 + ln, (av[m][1][r] + bv1v) * s);
            }
        }
    }
}

// ---------------------------------------------------------------------------
// Out projection + residual + LayerNorm. Block = 32 rows (two M-tiles), 8 waves.
// ---------------------------------------------------------------------------
__launch_bounds__(256)
__global__ void out_kernel(const float* __restrict__ x,
                           const float* __restrict__ agg,
                           const unsigned short* __restrict__ fw,
                           const float* __restrict__ bo,
                           const float* __restrict__ gamma,
                           const float* __restrict__ beta,
                           float* __restrict__ out, int n) {
    __shared__ __align__(16) unsigned short sA[32 * HIDDEN];
    __shared__ float sSum[8][32];
    __shared__ float sSum2[8][32];
    __shared__ float sMu[32], sRstd[32];

    int tid = threadIdx.x;
    int r0 = blockIdx.x * 32;

    {   // load+convert 32 agg rows (fp32 -> bf16) into LDS: thread t -> row t>>3, 32 vals
        int rr = r0 + (tid >> 3); if (rr >= n) rr = n - 1;
        const float* ap = agg + (size_t)rr * HIDDEN + (tid & 7) * 32;
        unsigned short* qd = sA + (tid >> 3) * HIDDEN + (tid & 7) * 32;
#pragma unroll
        for (int c = 0; c < 2; ++c) {
            FragU f;
#pragma unroll
            for (int j = 0; j < 16; ++j) f.h[j] = f2bf(ap[c * 16 + j]);
            uint4* q = (uint4*)(qd + c * 16);
            q[0] = f.u[0]; q[1] = f.u[1];
        }
    }
    __syncthreads();

    int lane = tid & 31;
    int wave = tid >> 5;
    int n0 = wave * 2, n1 = n0 + 1;
    int lg = (lane & 16) ? 8 : 0;
    int ln = lane & 15;

    v8f zero = {0, 0, 0, 0, 0, 0, 0, 0};
    v8f acc[2][2];                      // [mtile][ntile]
#pragma unroll
    for (int m = 0; m < 2; ++m) { acc[m][0] = zero; acc[m][1] = zero; }

#pragma unroll
    for (int kt = 0; kt < 8; ++kt) {
        FragU A0, A1;
        const unsigned short* ap0 = sA + ln * HIDDEN + kt * 32 + ((lane & 16) ? 16 : 0);
        A0.u[0] = *(const uint4*)ap0;
        A0.u[1] = *(const uint4*)(ap0 + 8);
        const unsigned short* ap1 = ap0 + 16 * HIDDEN;
        A1.u[0] = *(const uint4*)ap1;
        A1.u[1] = *(const uint4*)(ap1 + 8);

        size_t f0 = (size_t)(kt * 16 + n0) * 512 + lane * 16;
        size_t f1 = (size_t)(kt * 16 + n1) * 512 + lane * 16;
        FragU B0, B1;
        B0.u[0] = *(const uint4*)(fw + f0); B0.u[1] = *(const uint4*)(fw + f0 + 8);
        B1.u[0] = *(const uint4*)(fw + f1); B1.u[1] = *(const uint4*)(fw + f1 + 8);

        acc[0][0] = WMMA_BF16(A0.v, B0.v, acc[0][0]);
        acc[1][0] = WMMA_BF16(A1.v, B0.v, acc[1][0]);
        acc[0][1] = WMMA_BF16(A0.v, B1.v, acc[0][1]);
        acc[1][1] = WMMA_BF16(A1.v, B1.v, acc[1][1]);
    }

    int nA = n0 * 16 + ln, nB = n1 * 16 + ln;
    float boA = bo[nA], boB = bo[nB];
    float hA[2][8], hB[2][8];
#pragma unroll
    for (int m = 0; m < 2; ++m) {
#pragma unroll
        for (int r = 0; r < 8; ++r) {
            int row = m * 16 + r + lg;
            int gr = r0 + row; if (gr >= n) gr = n - 1;
            hA[m][r] = x[(size_t)gr * HIDDEN + nA] + acc[m][0][r] + boA;
            hB[m][r] = x[(size_t)gr * HIDDEN + nB] + acc[m][1][r] + boB;
            float s  = hA[m][r] + hB[m][r];
            float s2 = hA[m][r] * hA[m][r] + hB[m][r] * hB[m][r];
            s  += __shfl_xor(s, 1);  s  += __shfl_xor(s, 2);
            s  += __shfl_xor(s, 4);  s  += __shfl_xor(s, 8);
            s2 += __shfl_xor(s2, 1); s2 += __shfl_xor(s2, 2);
            s2 += __shfl_xor(s2, 4); s2 += __shfl_xor(s2, 8);
            if (ln == 0) { sSum[wave][row] = s; sSum2[wave][row] = s2; }
        }
    }
    __syncthreads();
    if (tid < 32) {
        float S = 0.f, S2 = 0.f;
#pragma unroll
        for (int w = 0; w < 8; ++w) { S += sSum[w][tid]; S2 += sSum2[w][tid]; }
        float mu = S * (1.0f / 256.0f);
        float var = S2 * (1.0f / 256.0f) - mu * mu;
        sMu[tid] = mu;
        sRstd[tid] = rsqrtf(var + 1e-5f);
    }
    __syncthreads();
    float gA = gamma[nA], gB = gamma[nB];
    float bA = beta[nA],  bB = beta[nB];
#pragma unroll
    for (int m = 0; m < 2; ++m) {
#pragma unroll
        for (int r = 0; r < 8; ++r) {
            int row = m * 16 + r + lg;
            int gr = r0 + row;
            if (gr < n) {
                float mu = sMu[row], rs = sRstd[row];
                out[(size_t)gr * HIDDEN + nA] = (hA[m][r] - mu) * rs * gA + bA;
                out[(size_t)gr * HIDDEN + nB] = (hB[m][r] - mu) * rs * gB + bB;
            }
        }
    }
}

// ---------------------------------------------------------------------------
// Host launcher
// ---------------------------------------------------------------------------
extern "C" void kernel_launch(void* const* d_in, const int* in_sizes, int n_in,
                              void* d_out, int out_size, void* d_ws, size_t ws_size,
                              hipStream_t stream) {
    (void)in_sizes; (void)n_in; (void)out_size; (void)ws_size;

    static const int NN[6]       = {2000, 30000, 5000, 100000, 10000, 10000}; // NODE_TYPES order
    static const int REL_SRC[10] = {0, 0, 0, 0, 2, 1, 1, 3, 3, 4};
    static const int REL_DST[10] = {1, 2, 3, 4, 3, 2, 4, 3, 3, 5};
    // position of each EDGE_TYPES rel / NODE_TYPES nt in alphabetically-sorted pytree order
    static const int REL_SORT[10] = {6, 5, 3, 4, 7, 9, 8, 1, 0, 2};
    static const int NT_SORT[6]   = {3, 5, 4, 1, 2, 0};
    const int E = ECNT;

    // pytree-flattened param base indices (sorted dict keys):
    //   edge_attn @26(x10), etw @36(x5: caption,cell,image,table,textblock),
    //   k @41(w,b x10), ln @61(g,b x6), out @73(w,b x6), q @85, v @105
    char* ws = (char*)d_ws;
    size_t cur = 0;
    auto alloc = [&](size_t b) { size_t o = cur; cur += (b + 255) & ~(size_t)255; return o; };

    size_t off_xb[6];
    for (int t = 0; t < 6; ++t) off_xb[t] = alloc((size_t)NN[t] * HIDDEN * 2);
    size_t off_fq[10], off_fk[10], off_fv[10];
    for (int r = 0; r < 10; ++r) {
        off_fq[r] = alloc(131072); off_fk[r] = alloc(131072); off_fv[r] = alloc(131072);
    }
    size_t off_fo[6] = {0};
    for (int t = 1; t < 6; ++t) off_fo[t] = alloc(131072);
    size_t aggStart = cur;
    size_t off_agg[6];
    for (int t = 0; t < 6; ++t) off_agg[t] = alloc((size_t)NN[t] * HIDDEN * 4);
    size_t aggBytes = cur - aggStart;
    size_t degStart = cur;
    size_t off_deg[10];
    for (int r = 0; r < 10; ++r) off_deg[r] = alloc((size_t)NN[REL_DST[r]] * 4);
    size_t degBytes = cur - degStart;
    size_t off_relw = alloc(64);

    // 1. relation mixing weights (softmax of etw)
    relw_kernel<<<1, 32, 0, stream>>>(
        (const float*)d_in[36], (const float*)d_in[37], (const float*)d_in[38],
        (const float*)d_in[39], (const float*)d_in[40], (float*)(ws + off_relw));

    // 2. node features -> bf16 (L2-resident gather source, ~80MB total)
    for (int t = 0; t < 6; ++t) {
        int groups = NN[t] * HIDDEN / 8;
        xcvt_kernel<<<(groups + 255) / 256, 256, 0, stream>>>(
            (const float*)d_in[t], (unsigned short*)(ws + off_xb[t]), groups);
    }

    // 3. weights -> bf16 WMMA B-fragments
    for (int r = 0; r < 10; ++r) {
        int sp = REL_SORT[r];
        wfrag_kernel<<<128, 32, 0, stream>>>((const float*)d_in[85 + 2 * sp],
                                             (unsigned short*)(ws + off_fq[r]));
        wfrag_kernel<<<128, 32, 0, stream>>>((const float*)d_in[41 + 2 * sp],
                                             (unsigned short*)(ws + off_fk[r]));
        wfrag_kernel<<<128, 32, 0, stream>>>((const float*)d_in[105 + 2 * sp],
                                             (unsigned short*)(ws + off_fv[r]));
    }
    for (int t = 1; t < 6; ++t)
        wfrag_kernel<<<128, 32, 0, stream>>>((const float*)d_in[73 + 2 * NT_SORT[t]],
                                             (unsigned short*)(ws + off_fo[t]));

    // 4. zero accumulators + degree counters
    hipMemsetAsync(ws + aggStart, 0, aggBytes, stream);
    hipMemsetAsync(ws + degStart, 0, degBytes, stream);

    // 5. degrees
    for (int r = 0; r < 10; ++r)
        deg_kernel<<<(E + 255) / 256, 256, 0, stream>>>(
            (const int*)d_in[7 + 2 * r], (float*)(ws + off_deg[r]), E);

    // 6. edge message passing (10 relations), 32 edges per block
    for (int r = 0; r < 10; ++r) {
        int sp = REL_SORT[r];
        edge_kernel<<<(E + 31) / 32, 256, 0, stream>>>(
            (const unsigned short*)(ws + off_xb[REL_SRC[r]]),
            (const int*)d_in[6 + 2 * r], (const int*)d_in[7 + 2 * r],
            (const unsigned short*)(ws + off_fq[r]),
            (const unsigned short*)(ws + off_fk[r]),
            (const unsigned short*)(ws + off_fv[r]),
            (const float*)d_in[86 + 2 * sp],   // bq
            (const float*)d_in[42 + 2 * sp],   // bk
            (const float*)d_in[106 + 2 * sp],  // bv
            (const float*)d_in[26 + sp],       // edge_attn
            (const float*)(ws + off_deg[r]),
            (const float*)(ws + off_relw) + r,
            (float*)(ws + off_agg[REL_DST[r]]), E);
    }

    // 7. per-node-type output
    size_t outOff[6];
    { size_t c = 0; for (int t = 0; t < 6; ++t) { outOff[t] = c; c += (size_t)NN[t] * HIDDEN; } }
    // query has no incoming relations: identity
    hipMemcpyAsync((float*)d_out + outOff[0], d_in[0],
                   (size_t)NN[0] * HIDDEN * sizeof(float),
                   hipMemcpyDeviceToDevice, stream);
    for (int t = 1; t < 6; ++t) {
        out_kernel<<<(NN[t] + 31) / 32, 256, 0, stream>>>(
            (const float*)d_in[t],
            (const float*)(ws + off_agg[t]),
            (const unsigned short*)(ws + off_fo[t]),
            (const float*)d_in[74 + 2 * NT_SORT[t]],   // bo
            (const float*)d_in[61 + 2 * NT_SORT[t]],   // gamma
            (const float*)d_in[62 + 2 * NT_SORT[t]],   // beta
            (float*)d_out + outOff[t], NN[t]);
    }
}
